// HModule_13786845020429
// MI455X (gfx1250) — compile-verified, matrix operand order checked by reading, not run on hardware
//
#include <hip/hip_runtime.h>
#include <hip/hip_bf16.h>

// ---------------------------------------------------------------------------
// MI455X (gfx1250) transformer block: add+rmsnorm -> qkv GEMM -> rope ->
// flash attention -> proj GEMM (+resid) -> rmsnorm -> ffn1 GEMM (+silu) ->
// ffn2 GEMM (+resid). GEMMs: WMMA f16 with async Global->LDS double-buffered
// tiles (ASYNCcnt path). Attention: fused flash kernel, WMMA for QK^T and PV.
// ---------------------------------------------------------------------------

typedef __attribute__((ext_vector_type(16))) _Float16 v16h;
typedef __attribute__((ext_vector_type(8)))  _Float16 v8h;
typedef __attribute__((ext_vector_type(8)))  float    v8f;

#define BB 2
#define SS 2048
#define DD 1024
#define FF 4096
#define HH 16
#define HD 64
#define MM (BB*SS)   // 4096 rows

// WMMA wrapper: D = A(16x32 f16) x B(32x16 f16) + C(16x16 f32)
__device__ __forceinline__ v8f wmma16x16x32(v16h a, v16h b, v8f c) {
    return __builtin_amdgcn_wmma_f32_16x16x32_f16(
        /*neg_a=*/false, a, /*neg_b=*/false, b,
        /*c_mod=*/(short)0, c, /*reuse_a=*/false, /*reuse_b=*/false);
}

// Fragment load matching the documented 16-bit 16x32 A layout (B is the
// column-major mirror): lane&15 = row (A) / col (B); lane>=16 picks the
// K+8 / K+24 half-blocks.  `base` points at (row-start + k0), khalf =
// (lane>>4)*8.  Two contiguous 16-byte loads per fragment.
__device__ __forceinline__ v16h load_frag(const _Float16* base, int khalf) {
    v8h lo = *(const v8h*)(base + khalf);
    v8h hi = *(const v8h*)(base + khalf + 16);
    v16h f;
#pragma unroll
    for (int i = 0; i < 8; ++i) { f[i] = lo[i]; f[8 + i] = hi[i]; }
    return f;
}

// Async Global->LDS copy, 16 bytes per lane (GVS mode: SGPR base + per-lane
// 32-bit byte offset). Tracked by ASYNCcnt.
__device__ __forceinline__ void async_cp_b128(unsigned lds_addr, unsigned goff,
                                              const _Float16* base) {
    asm volatile("global_load_async_to_lds_b128 %0, %1, %2"
                 :: "v"(lds_addr), "v"(goff), "s"(base) : "memory");
}
__device__ __forceinline__ unsigned lds_off(const void* p) {
    return (unsigned)(uintptr_t)p;   // LDS aperture: low 32 bits = LDS address
}

// ---------------------------------------------------------------------------
// Fused (a [+ b]) -> rmsnorm: writes optional f32 residual copy and f16 norm.
// ---------------------------------------------------------------------------
__global__ __launch_bounds__(256) void add_rmsnorm_kernel(
    const float* __restrict__ a, const float* __restrict__ b,
    const float* __restrict__ g, float* __restrict__ x_out,
    _Float16* __restrict__ h_out)
{
    __shared__ float red[8];
    __shared__ float s_inv;
    const int row = blockIdx.x;
    const size_t base = (size_t)row * DD;
    float ss = 0.f;
    for (int i = threadIdx.x; i < DD; i += 256) {
        float v = a[base + i] + (b ? b[base + i] : 0.f);
        ss += v * v;
        if (x_out) x_out[base + i] = v;
    }
#pragma unroll
    for (int off = 16; off > 0; off >>= 1) ss += __shfl_down(ss, off, 32);
    if ((threadIdx.x & 31) == 0) red[threadIdx.x >> 5] = ss;
    __syncthreads();
    if (threadIdx.x == 0) {
        float t = 0.f;
#pragma unroll
        for (int i = 0; i < 8; ++i) t += red[i];
        s_inv = rsqrtf(t / (float)DD + 1e-6f);
    }
    __syncthreads();
    const float inv = s_inv;
    for (int i = threadIdx.x; i < DD; i += 256) {
        float v = x_out ? x_out[base + i] : (a[base + i] + (b ? b[base + i] : 0.f));
        h_out[base + i] = (_Float16)(v * inv * g[i]);
    }
}

// ---------------------------------------------------------------------------
// Weight conversion: w (K x N, f32, row-major) -> wt (N x K, f16).
// WMMA's B operand wants column-major (contiguous K per column).
// ---------------------------------------------------------------------------
__global__ __launch_bounds__(256) void convert_wt_kernel(
    const float* __restrict__ w, _Float16* __restrict__ wt, int Kd, int N)
{
    __shared__ float tile[32][33];
    const int kb = blockIdx.x * 32, nb = blockIdx.y * 32;
    const int tx = threadIdx.x & 31, ty = threadIdx.x >> 5;
#pragma unroll
    for (int i = ty; i < 32; i += 8)
        tile[i][tx] = w[(size_t)(kb + i) * N + nb + tx];
    __syncthreads();
#pragma unroll
    for (int i = ty; i < 32; i += 8)
        wt[(size_t)(nb + i) * Kd + kb + tx] = (_Float16)tile[tx][i];
}

// ---------------------------------------------------------------------------
// WMMA GEMM with async Global->LDS double-buffered tiles.
// C(MxN,f32) = A(MxK,f16 row-major) x Bt(NxK,f16)^T.
// Block tile 128x64, 8 waves each owning 32x32 (2x2 WMMA accumulators).
// Per K-step (32): stage A(128x32)+B(64x32) = 12 KB once per block via
// ASYNCcnt-tracked copies (3 instr/wave), fragments read from LDS.
// mode 0: f32 store | mode 1: f32 resid+acc | mode 2: f16 silu(acc).
// ---------------------------------------------------------------------------
__global__ __launch_bounds__(256) void gemm_wmma_kernel(
    const _Float16* __restrict__ A, const _Float16* __restrict__ Bt,
    int N, int Kd, int mode, const float* __restrict__ resid,
    float* __restrict__ outf, _Float16* __restrict__ outh)
{
    __shared__ _Float16 sA[2][128 * 32];   // 2 x 8 KB
    __shared__ _Float16 sB[2][64 * 32];    // 2 x 4 KB

    const int tid  = threadIdx.x;
    const int wave = tid >> 5, lane = tid & 31;
    const int l16 = lane & 15, khalf = (lane >> 4) * 8;
    const int wm = wave & 3, wn = wave >> 2;       // 4 waves in M, 2 in N
    const int blockRow0 = blockIdx.y * 128;
    const int blockCol0 = blockIdx.x * 64;
    const int rowBase = blockRow0 + wm * 32;
    const int colBase = blockCol0 + wn * 32;

    // Copy assignment: 16-byte chunks. A: 128 rows x 4 chunks = 512 (2/thread),
    // B: 64 rows x 4 chunks = 256 (1/thread).
    const int seg  = tid & 3;            // which 16B of the 64B row
    const int crow = tid >> 2;           // 0..63
    const unsigned aoff0 = (unsigned)(((size_t)(blockRow0 + crow)      * Kd + seg * 8) * 2);
    const unsigned aoff1 = (unsigned)(((size_t)(blockRow0 + crow + 64) * Kd + seg * 8) * 2);
    const unsigned boff  = (unsigned)(((size_t)(blockCol0 + crow)      * Kd + seg * 8) * 2);
    unsigned ldsA0[2], ldsA1[2], ldsB0[2];
#pragma unroll
    for (int s = 0; s < 2; ++s) {
        ldsA0[s] = lds_off(&sA[s][(crow)      * 32 + seg * 8]);
        ldsA1[s] = lds_off(&sA[s][(crow + 64) * 32 + seg * 8]);
        ldsB0[s] = lds_off(&sB[s][(crow)      * 32 + seg * 8]);
    }

    v8f acc00 = {}, acc01 = {}, acc10 = {}, acc11 = {};

    // Prologue: stage first tile into buffer 0.
    async_cp_b128(ldsA0[0], aoff0, A);
    async_cp_b128(ldsA1[0], aoff1, A);
    async_cp_b128(ldsB0[0], boff,  Bt);

    int buf = 0;
    for (int k0 = 0; k0 < Kd; k0 += 32, buf ^= 1) {
        const bool more = (k0 + 32) < Kd;
        if (more) {   // stage next tile into the other buffer
            const unsigned kb = (unsigned)((k0 + 32) * 2);
            async_cp_b128(ldsA0[buf ^ 1], aoff0 + kb, A);
            async_cp_b128(ldsA1[buf ^ 1], aoff1 + kb, A);
            async_cp_b128(ldsB0[buf ^ 1], boff  + kb, Bt);
            asm volatile("s_wait_asynccnt 0x3" ::: "memory");  // current buf done
        } else {
            asm volatile("s_wait_asynccnt 0x0" ::: "memory");
        }
        __syncthreads();   // all waves' copies visible

        const _Float16* a0 = &sA[buf][(wm * 32 + 0  + l16) * 32];
        const _Float16* a1 = &sA[buf][(wm * 32 + 16 + l16) * 32];
        const _Float16* b0 = &sB[buf][(wn * 32 + 0  + l16) * 32];
        const _Float16* b1 = &sB[buf][(wn * 32 + 16 + l16) * 32];
        v16h af0 = load_frag(a0, khalf);
        v16h af1 = load_frag(a1, khalf);
        v16h bf0 = load_frag(b0, khalf);
        v16h bf1 = load_frag(b1, khalf);
        acc00 = wmma16x16x32(af0, bf0, acc00);
        acc01 = wmma16x16x32(af0, bf1, acc01);
        acc10 = wmma16x16x32(af1, bf0, acc10);
        acc11 = wmma16x16x32(af1, bf1, acc11);

        __syncthreads();   // reads done before next stage overwrites buf
    }

    const int mrow = (lane >> 4) * 8;   // C/D layout: M = r + 8*(lane>=16)
#pragma unroll
    for (int r = 0; r < 8; ++r) {
        const int rows[2] = { rowBase + mrow + r, rowBase + 16 + mrow + r };
        const int cols[2] = { colBase + l16,      colBase + 16 + l16 };
        const float vals[2][2] = { { acc00[r], acc01[r] }, { acc10[r], acc11[r] } };
#pragma unroll
        for (int i = 0; i < 2; ++i)
#pragma unroll
            for (int j = 0; j < 2; ++j) {
                const size_t idx = (size_t)rows[i] * N + cols[j];
                const float v = vals[i][j];
                if (mode == 0)      outf[idx] = v;
                else if (mode == 1) outf[idx] = resid[idx] + v;
                else {
                    const float sg = 1.f / (1.f + __expf(-v));
                    outh[idx] = (_Float16)(v * sg);
                }
            }
    }
}

// ---------------------------------------------------------------------------
// RoPE + qkv split: qkv_f32 (B,S,3,H,64) -> q/k (roped, f16) & v (f16),
// all in (B,H,S,64) layout. One thread per (b,h,s,dpair<32).
// ---------------------------------------------------------------------------
__global__ __launch_bounds__(256) void rope_split_kernel(
    const float* __restrict__ qkv, _Float16* __restrict__ q,
    _Float16* __restrict__ k, _Float16* __restrict__ v)
{
    const size_t idx = (size_t)blockIdx.x * blockDim.x + threadIdx.x;
    const int dp = (int)(idx & 31);
    size_t t = idx >> 5;
    const int s = (int)(t % SS); t /= SS;
    const int h = (int)(t % HH);
    const int b = (int)(t / HH);

    const size_t rowbase = ((size_t)b * SS + s) * (3 * HH * HD);
    const float* qp = qkv + rowbase + 0 * HH * HD + h * HD;
    const float* kp = qkv + rowbase + 1 * HH * HD + h * HD;
    const float* vp = qkv + rowbase + 2 * HH * HD + h * HD;

    // inv_freq = 10000^(-dp/32) = exp(-dp * ln(10000)/32)
    const float freq = (float)s * __expf(-(float)dp * (9.210340371976184f / 32.f));
    const float c = __cosf(freq), sn = __sinf(freq);

    const size_t obase = (((size_t)b * HH + h) * SS + s) * HD;
    const float q1 = qp[dp], q2 = qp[dp + 32];
    q[obase + dp]      = (_Float16)(q1 * c - q2 * sn);
    q[obase + dp + 32] = (_Float16)(q2 * c + q1 * sn);
    const float k1 = kp[dp], k2 = kp[dp + 32];
    k[obase + dp]      = (_Float16)(k1 * c - k2 * sn);
    k[obase + dp + 32] = (_Float16)(k2 * c + k1 * sn);
    v[obase + dp]      = (_Float16)vp[dp];
    v[obase + dp + 32] = (_Float16)vp[dp + 32];
}

// ---------------------------------------------------------------------------
// Flash attention: grid (S/64 query tiles, B*H). Block = 256 thr = 8 waves.
// Each block: 64 query rows, loops over 32 key tiles of 64.
//   phase A: S = Q K^T * 0.125 via WMMA -> LDS (f32)
//   phase B: online softmax per row (threads 0..63), P -> LDS (f16)
//   phase C: O = O*alpha + P x V^T via WMMA (V transposed through LDS)
// Output: (B,S,H*64) f16, ready as GEMM A-operand for the proj matmul.
// ---------------------------------------------------------------------------
__global__ __launch_bounds__(256) void flash_attn_kernel(
    const _Float16* __restrict__ Q, const _Float16* __restrict__ K,
    const _Float16* __restrict__ V, _Float16* __restrict__ O)
{
    __shared__ float    s_sc[64 * 64];      // raw scores (16 KB)
    __shared__ _Float16 s_p [64 * 64];      // softmax numerators (8 KB)
    __shared__ _Float16 s_vt[64 * 64];      // V tile transposed [d][s] (8 KB)
    __shared__ float    s_M[64], s_L[64], s_alpha[64];

    const int bh    = blockIdx.y;
    const int qtile = blockIdx.x;
    const int tid   = threadIdx.x;
    const int wave  = tid >> 5, lane = tid & 31;
    const int l16 = lane & 15, khalf = (lane >> 4) * 8;
    const int mi = wave & 3;                // 16-row M strip
    const int nj = wave >> 2;               // covers N tiles 2*nj, 2*nj+1

    const _Float16* Qb = Q + ((size_t)bh * SS + qtile * 64) * HD;
    const _Float16* Kb = K + (size_t)bh * SS * HD;
    const _Float16* Vb = V + (size_t)bh * SS * HD;

    if (tid < 64) { s_M[tid] = -1e30f; s_L[tid] = 0.f; }

    // Q fragments for this wave's M strip (K = 64 -> 2 fragments), kept live.
    const v16h qf0 = load_frag(Qb + (size_t)(mi * 16 + l16) * HD + 0,  khalf);
    const v16h qf1 = load_frag(Qb + (size_t)(mi * 16 + l16) * HD + 32, khalf);

    v8f o0 = {}, o1 = {};
    __syncthreads();

    for (int kt = 0; kt < SS / 64; ++kt) {
        const int key0 = kt * 64;

        // Load V tile (64x64) transposed into LDS.
        {
            const int srow = tid >> 2;
            const int d0 = (tid & 3) * 16;
            const _Float16* vp = Vb + (size_t)(key0 + srow) * HD + d0;
#pragma unroll
            for (int i = 0; i < 16; ++i)
                s_vt[(d0 + i) * 64 + srow] = vp[i];
        }

        // Phase A: score subtiles (each wave: 2 x 16x16).
#pragma unroll
        for (int t = 0; t < 2; ++t) {
            const int ni = 2 * nj + t;
            const _Float16* Krow = Kb + (size_t)(key0 + ni * 16 + l16) * HD;
            v16h kf0 = load_frag(Krow + 0,  khalf);
            v16h kf1 = load_frag(Krow + 32, khalf);
            v8f acc = {};
            acc = wmma16x16x32(qf0, kf0, acc);
            acc = wmma16x16x32(qf1, kf1, acc);
            const int mrow = mi * 16 + (lane >> 4) * 8;
#pragma unroll
            for (int r = 0; r < 8; ++r)
                s_sc[(mrow + r) * 64 + ni * 16 + l16] = acc[r] * 0.125f;
        }
        __syncthreads();

        // Phase B: online softmax, one thread per query row.
        if (tid < 64) {
            const float m_old = s_M[tid];
            float tmax = m_old;
            const float* row = &s_sc[tid * 64];
            for (int j = 0; j < 64; ++j) tmax = fmaxf(tmax, row[j]);
            const float alpha = __expf(m_old - tmax);
            float sum = 0.f;
            _Float16* prow = &s_p[tid * 64];
            for (int j = 0; j < 64; ++j) {
                const float p = __expf(row[j] - tmax);
                sum += p;
                prow[j] = (_Float16)p;
            }
            s_M[tid] = tmax;
            s_L[tid] = s_L[tid] * alpha + sum;
            s_alpha[tid] = alpha;
        }
        __syncthreads();

        // Phase C: rescale O, accumulate P x V^T.
        {
            const int mrow = mi * 16 + (lane >> 4) * 8;
#pragma unroll
            for (int r = 0; r < 8; ++r) {
                const float a = s_alpha[mrow + r];
                o0[r] *= a; o1[r] *= a;
            }
            v16h pf0 = load_frag(&s_p[(mi * 16 + l16) * 64 + 0],  khalf);
            v16h pf1 = load_frag(&s_p[(mi * 16 + l16) * 64 + 32], khalf);
            {
                const int d0 = (2 * nj + 0) * 16 + l16;
                v16h vf0 = load_frag(&s_vt[d0 * 64 + 0],  khalf);
                v16h vf1 = load_frag(&s_vt[d0 * 64 + 32], khalf);
                o0 = wmma16x16x32(pf0, vf0, o0);
                o0 = wmma16x16x32(pf1, vf1, o0);
            }
            {
                const int d1 = (2 * nj + 1) * 16 + l16;
                v16h vf0 = load_frag(&s_vt[d1 * 64 + 0],  khalf);
                v16h vf1 = load_frag(&s_vt[d1 * 64 + 32], khalf);
                o1 = wmma16x16x32(pf0, vf0, o1);
                o1 = wmma16x16x32(pf1, vf1, o1);
            }
        }
        __syncthreads();   // protect s_sc/s_p/s_vt for next tile
    }

    // Final: O /= L, store to (B,S,H*64) f16.
    const int b = bh / HH, h = bh % HH;
    const int mrow = mi * 16 + (lane >> 4) * 8;
#pragma unroll
    for (int r = 0; r < 8; ++r) {
        const int qrow = qtile * 64 + mrow + r;
        const float inv = 1.f / s_L[mrow + r];
        const int d0 = (2 * nj + 0) * 16 + l16;
        const int d1 = (2 * nj + 1) * 16 + l16;
        const size_t base = ((size_t)(b * SS + qrow)) * (HH * HD) + h * HD;
        O[base + d0] = (_Float16)(o0[r] * inv);
        O[base + d1] = (_Float16)(o1[r] * inv);
    }
}

// ---------------------------------------------------------------------------
// Host-side launcher.
// ---------------------------------------------------------------------------
extern "C" void kernel_launch(void* const* d_in, const int* in_sizes, int n_in,
                              void* d_out, int out_size, void* d_ws, size_t ws_size,
                              hipStream_t stream) {
    const float* zH    = (const float*)d_in[0];
    const float* zL    = (const float*)d_in[1];
    const float* wqkv  = (const float*)d_in[2];   // (D, 3D)
    const float* wproj = (const float*)d_in[3];   // (D, D)
    const float* wffn1 = (const float*)d_in[4];   // (D, F)
    const float* wffn2 = (const float*)d_in[5];   // (F, D)
    const float* g1    = (const float*)d_in[6];
    const float* g2    = (const float*)d_in[7];
    float* out = (float*)d_out;

    // Flat workspace layout (~168 MB total).
    char* w = (char*)d_ws;
    float*     x_f32   = (float*)    (w + ((size_t)0   << 20));  // 16 MB  x = zH+zL
    float*     x2_f32  = (float*)    (w + ((size_t)16  << 20));  // 16 MB  x + attn@proj
    _Float16*  hid     = (_Float16*) (w + ((size_t)32  << 20));  //  8 MB  rmsnorm1 f16
    _Float16*  h2      = (_Float16*) (w + ((size_t)40  << 20));  //  8 MB  rmsnorm2 f16
    _Float16*  attn    = (_Float16*) (w + ((size_t)48  << 20));  //  8 MB  attention out
    _Float16*  qh      = (_Float16*) (w + ((size_t)56  << 20));  //  8 MB  q (roped)
    _Float16*  kh      = (_Float16*) (w + ((size_t)64  << 20));  //  8 MB  k (roped)
    _Float16*  vh      = (_Float16*) (w + ((size_t)72  << 20));  //  8 MB  v
    _Float16*  ffn1o   = (_Float16*) (w + ((size_t)80  << 20));  // 32 MB  silu(h@w1)
    float*     qkv     = (float*)    (w + ((size_t)112 << 20));  // 48 MB  qkv f32
    _Float16*  wT      = (_Float16*) (w + ((size_t)160 << 20));  //  8 MB  weight (reused)

    const dim3 blk(256);

    // 1) x = zH + zL ; hid = rmsnorm(x, g1) as f16
    add_rmsnorm_kernel<<<dim3(MM), blk, 0, stream>>>(zH, zL, g1, x_f32, hid);

    // 2) qkv = hid @ w_qkv            (M=4096, K=1024, N=3072)
    convert_wt_kernel<<<dim3(DD/32, 3*DD/32), blk, 0, stream>>>(wqkv, wT, DD, 3*DD);
    gemm_wmma_kernel<<<dim3(3*DD/64, MM/128), blk, 0, stream>>>(
        hid, wT, 3*DD, DD, 0, nullptr, qkv, nullptr);

    // 3) rope(q), rope(k), split v -> (B,H,S,64) f16
    rope_split_kernel<<<dim3((BB*HH*SS*32)/256), blk, 0, stream>>>(qkv, qh, kh, vh);

    // 4) flash attention -> attn (B,S,H*64) f16
    flash_attn_kernel<<<dim3(SS/64, BB*HH), blk, 0, stream>>>(qh, kh, vh, attn);

    // 5) x2 = x + attn @ w_proj       (M=4096, K=1024, N=1024)
    convert_wt_kernel<<<dim3(DD/32, DD/32), blk, 0, stream>>>(wproj, wT, DD, DD);
    gemm_wmma_kernel<<<dim3(DD/64, MM/128), blk, 0, stream>>>(
        attn, wT, DD, DD, 1, x_f32, x2_f32, nullptr);

    // 6) h2 = rmsnorm(x2, g2) as f16
    add_rmsnorm_kernel<<<dim3(MM), blk, 0, stream>>>(x2_f32, nullptr, g2, nullptr, h2);

    // 7) ffn1o = silu(h2 @ w_ffn1) f16  (M=4096, K=1024, N=4096)
    convert_wt_kernel<<<dim3(DD/32, FF/32), blk, 0, stream>>>(wffn1, wT, DD, FF);
    gemm_wmma_kernel<<<dim3(FF/64, MM/128), blk, 0, stream>>>(
        h2, wT, FF, DD, 2, nullptr, nullptr, ffn1o);

    // 8) out = x2 + ffn1o @ w_ffn2    (M=4096, K=4096, N=1024)
    convert_wt_kernel<<<dim3(FF/32, DD/32), blk, 0, stream>>>(wffn2, wT, FF, DD);
    gemm_wmma_kernel<<<dim3(DD/64, MM/128), blk, 0, stream>>>(
        ffn1o, wT, DD, FF, 1, x2_f32, out, nullptr);
}
